// MVGRL_57878979281252
// MI455X (gfx1250) — matrix operand order repaired and checked by reading, not compiled, not run on hardware
//
#include <hip/hip_runtime.h>

// ---------------------------------------------------------------------------
// MVGRL forward for MI455X (gfx1250, wave32).
// GEMMs: v_wmma_f32_16x16x32_f16 with all A-inputs pre-converted to f16
// (producers emit f16 directly), so the inner loop is pure loads + WMMA:
// 2x global_load_b128 for the A fragment, 2x ds_load_b128 per B fragment
// from LDS-staged transposed weights, software-pipelined one tile ahead.
// Round-5 disasm showed in-loop fp32->f16 conversion inflating register
// pressure and serializing the A loads. No conditional loads (addresses
// clamped, stores guarded): EXEC is all-ones at every v_wmma. Edge
// aggregation: global_atomic_add_f32 on L2-resident feature buffers.
// ---------------------------------------------------------------------------

typedef __attribute__((ext_vector_type(16))) _Float16 v16h;
typedef __attribute__((ext_vector_type(8)))  _Float16 v8h;
typedef __attribute__((ext_vector_type(8)))  float    v8f;

// ---------------- elementwise / utility kernels ----------------------------

__global__ void zero_kernel(float* p, long long n) {
  long long i = blockIdx.x * (long long)blockDim.x + threadIdx.x;
  if (i < n) p[i] = 0.f;
}

__global__ void deg_kernel(const int* __restrict__ dst, float* __restrict__ deg, long long E) {
  long long e = blockIdx.x * (long long)blockDim.x + threadIdx.x;
  if (e < E) atomicAdd(&deg[dst[e]], 1.0f);
}

__global__ void rsqrt_kernel(float* d, long long n) {
  long long i = blockIdx.x * (long long)blockDim.x + threadIdx.x;
  if (i < n) { float v = d[i]; d[i] = (v > 0.f) ? rsqrtf(v) : 0.f; }
}

__global__ void fill_bias_kernel(float* __restrict__ out, const float* __restrict__ bias,
                                 long long n, int fmask) {
  long long i = blockIdx.x * (long long)blockDim.x + threadIdx.x;
  if (i < n) out[i] = bias[(int)i & fmask];
}

__global__ void relu_kernel(float* p, long long n) {
  long long i = blockIdx.x * (long long)blockDim.x + threadIdx.x;
  if (i < n) p[i] = fmaxf(p[i], 0.f);
}

// c16 = f16(a + b)
__global__ void add16_kernel(const float* __restrict__ a, const float* __restrict__ b,
                             _Float16* __restrict__ c, long long n) {
  long long i = blockIdx.x * (long long)blockDim.x + threadIdx.x;
  if (i < n) c[i] = (_Float16)(a[i] + b[i]);
}

// straight fp32 -> f16 conversion (same layout)
__global__ void cvt16_kernel(const float* __restrict__ a, _Float16* __restrict__ c, long long n) {
  long long i = blockIdx.x * (long long)blockDim.x + threadIdx.x;
  if (i < n) c[i] = (_Float16)a[i];
}

// Transpose fp32 W[K x N] (row-major) -> f16 Wt[N x K] (row-major, i.e.
// column-major of W). Makes B fragments contiguous 32B per lane.
__global__ void transpose_f16_kernel(const float* __restrict__ W, _Float16* __restrict__ Wt,
                                     int K, int N) {
  int i = blockIdx.x * blockDim.x + threadIdx.x;
  if (i < K * N) {
    int k = i / N, n = i - k * N;
    Wt[(size_t)n * K + k] = (_Float16)W[i];
  }
}

// ---------------- GCN edge propagation (scatter-add, one wave per edge) ----

__global__ void propagate_kernel(const float* __restrict__ h, const int* __restrict__ src,
                                 const int* __restrict__ dst, const float* __restrict__ dinv,
                                 float* __restrict__ out, long long E) {
  long long gid = blockIdx.x * (long long)blockDim.x + threadIdx.x;
  long long e = gid >> 5;                 // one wave32 per edge
  if (e >= E) return;
  int lane = (int)(gid & 31);             // 32 lanes x float4 = 128 features
  int s = src[e];
  int d = dst[e];
  float w = dinv[s] * dinv[d];
  long long ep = e + 64;                  // prefetch a future source row
  if (ep < E) {
    int sp = src[ep];
    __builtin_prefetch(h + (size_t)sp * 128 + lane * 4, 0, 1);
  }
  const float4 hv = *reinterpret_cast<const float4*>(h + (size_t)s * 128 + lane * 4);
  float* o = out + (size_t)d * 128 + lane * 4;
  atomicAdd(o + 0, hv.x * w);
  atomicAdd(o + 1, hv.y * w);
  atomicAdd(o + 2, hv.z * w);
  atomicAdd(o + 3, hv.w * w);
}

// ---------------- BatchNorm stats / normalize ------------------------------

__global__ void col_stats_kernel(const float* __restrict__ X, float* __restrict__ stats, int N) {
  int j = threadIdx.x;                    // blockDim = 128 (thread = column)
  int r0 = blockIdx.x * 256;
  int rend = r0 + 256; if (rend > N) rend = N;
  float s = 0.f, sq = 0.f;
  for (int r = r0; r < rend; ++r) {
    float v = X[(size_t)r * 128 + j];
    s += v; sq += v * v;
  }
  atomicAdd(&stats[j], s);
  atomicAdd(&stats[128 + j], sq);
}

// normalize fp32 X -> f16 Y (BN output only feeds the next GEMM)
__global__ void bn16_kernel(const float* __restrict__ X, const float* __restrict__ stats,
                            const float* __restrict__ g, const float* __restrict__ bt,
                            _Float16* __restrict__ Y, long long n, float invN) {
  long long i = blockIdx.x * (long long)blockDim.x + threadIdx.x;
  if (i >= n) return;
  int j = (int)i & 127;
  float mu  = stats[j] * invN;
  float var = stats[128 + j] * invN - mu * mu;   // biased variance
  float x = X[i];
  Y[i] = (_Float16)(g[j] * (x - mu) * rsqrtf(var + 1e-5f) + bt[j]);
}

__global__ void colsum_kernel(const float* __restrict__ X, float* __restrict__ s, int N) {
  int j = threadIdx.x;
  int r0 = blockIdx.x * 256;
  int rend = r0 + 256; if (rend > N) rend = N;
  float acc = 0.f;
  for (int r = r0; r < rend; ++r) acc += X[(size_t)r * 128 + j];
  atomicAdd(&s[j], acc);
}

// ---------------- tiny 1x128 MLP for graph summaries -----------------------

__global__ void tiny_mlp_kernel(const float* __restrict__ s,
                                const float* __restrict__ W1, const float* __restrict__ b1,
                                const float* __restrict__ W2, const float* __restrict__ b2,
                                float* __restrict__ out) {
  __shared__ float sv[128];
  __shared__ float t[128];
  int j = threadIdx.x;
  sv[j] = s[j];
  __syncthreads();
  float acc = b1[j];
  for (int k = 0; k < 128; ++k) acc += sv[k] * W1[k * 128 + j];
  t[j] = fmaxf(acc, 0.f);
  __syncthreads();
  float acc2 = b2[j];
  for (int k = 0; k < 128; ++k) acc2 += t[k] * W2[k * 128 + j];
  out[j] = acc2;
}

// ---------------- WMMA strip GEMM with LDS-staged weights ------------------
// C[M x Nc] = A[M x KT] @ B[KT x Nc] (+bias, optional relu), Nc <= NT*16.
// A is f16 row-major; Bt is B pre-transposed to f16 (Bt[n*KT + k]) and
// staged in LDS once per workgroup (<= 64KB of the WGP's 320KB). One wave
// computes a 16 x (16*NT) strip. Inner loop is pure data movement + WMMA:
// A fragment = 2x global_load_b128, B fragments rotate through a 2-deep
// software pipeline of ds_load_b128 pairs. Output stored as fp32 or f16.

template <int NT, int KT>
__launch_bounds__(256)
__global__ void gemm_strip_kernel(const _Float16* __restrict__ A, const _Float16* __restrict__ Bt,
                                  const float* __restrict__ bias, void* __restrict__ Cout,
                                  int M, int Ncols, int do_relu, int f16out) {
  __shared__ __align__(32) _Float16 bs[NT * 16 * KT];

  // cooperative stage of Bt into LDS (16B chunks); Ncols*KT is a multiple of 8
  {
    const uint4* gsrc = reinterpret_cast<const uint4*>(Bt);
    uint4* ldst = reinterpret_cast<uint4*>(bs);
    const int nvec = Ncols * KT / 8;           // 8 halfs per uint4
    for (int i = (int)threadIdx.x; i < nvec; i += (int)blockDim.x) ldst[i] = gsrc[i];
  }
  __syncthreads();

  const int m_tiles = (M + 15) >> 4;
  const int wid = blockIdx.x * (blockDim.x >> 5) + ((int)threadIdx.x >> 5);
  if (wid < m_tiles) {                         // wave-uniform guard
    const int lane = (int)threadIdx.x & 31;
    const int grp  = lane >> 4;                // 0: lanes 0-15, 1: lanes 16-31
    const int lr   = lane & 15;
    const int m0   = wid << 4;

    int mrow = m0 + lr; if (mrow >= M) mrow = M - 1;   // clamp; stores guarded
    const _Float16* arow = A + (size_t)mrow * KT;

    const _Float16* brow[NT];
    #pragma unroll
    for (int t = 0; t < NT; ++t) {
      int n = t * 16 + lr; if (n >= Ncols) n = Ncols - 1;  // clamp; stores guarded
      brow[t] = bs + n * KT;
    }

    v8f acc[NT];
    #pragma unroll
    for (int t = 0; t < NT; ++t) { v8f z = {}; acc[t] = z; }

    for (int k0 = 0; k0 < KT; k0 += 32) {
      // A fragment (16x32 f16): one contiguous 16-half run/lane (k = 16*grp + s),
      // matching the B fragment k order.
      v16h a = *reinterpret_cast<const v16h*>(arow + k0 + grp * 16);

      // 2-deep software pipeline over column tiles: load B for t+1 before
      // the WMMA for t (only 2 fragments live -> low register pressure).
      v16h bcur = *reinterpret_cast<const v16h*>(brow[0] + k0 + grp * 16);
      #pragma unroll
      for (int t = 0; t < NT; ++t) {
        v16h bnext;
        if (t + 1 < NT)
          bnext = *reinterpret_cast<const v16h*>(brow[t + 1] + k0 + grp * 16);
        acc[t] = __builtin_amdgcn_wmma_f32_16x16x32_f16(
            /*neg_a=*/false, a, /*neg_b=*/false, bcur,
            /*c_mod=*/(short)0, acc[t], /*reuse_a=*/false, /*reuse_b=*/false);
        if (t + 1 < NT) bcur = bnext;
      }
    }

    #pragma unroll
    for (int t = 0; t < NT; ++t) {
      int n = t * 16 + lr;
      int nb = n < Ncols ? n : Ncols - 1;              // clamped bias index
      float bb = bias ? bias[nb] : 0.f;
      bool nvalid = (n < Ncols);
      #pragma unroll
      for (int v = 0; v < 8; ++v) {
        int mm = m0 + grp * 8 + v;                     // C/D: M = half*8 + vgpr
        if (mm < M && nvalid) {
          float r = acc[t][v] + bb;
          if (do_relu) r = fmaxf(r, 0.f);
          if (f16out) ((_Float16*)Cout)[(size_t)mm * Ncols + n] = (_Float16)r;
          else        ((float*)Cout)[(size_t)mm * Ncols + n] = r;
        }
      }
    }
  }
}

// ---------------------------------------------------------------------------

static inline unsigned blk(long long n, int b) { return (unsigned)((n + b - 1) / b); }

static inline void launch_gemm(const _Float16* A, const _Float16* Bt, const float* bias,
                               void* C, int M, int K, int Nc, int relu, int f16out,
                               hipStream_t s) {
  int m_tiles = (M + 15) / 16;
  int blocks = (m_tiles + 7) / 8;                // 8 waves (row strips) per block
  if (K == 256 && Nc == 128)
    gemm_strip_kernel<8, 256><<<blocks, 256, 0, s>>>(A, Bt, bias, C, M, Nc, relu, f16out);
  else if (K == 128 && Nc == 128)
    gemm_strip_kernel<8, 128><<<blocks, 256, 0, s>>>(A, Bt, bias, C, M, Nc, relu, f16out);
  else if (K == 128 && Nc == 64)
    gemm_strip_kernel<4, 128><<<blocks, 256, 0, s>>>(A, Bt, bias, C, M, Nc, relu, f16out);
  else // K == 64, Nc == 40
    gemm_strip_kernel<3, 64><<<blocks, 256, 0, s>>>(A, Bt, bias, C, M, Nc, relu, f16out);
}

extern "C" void kernel_launch(void* const* d_in, const int* in_sizes, int n_in,
                              void* d_out, int out_size, void* d_ws, size_t ws_size,
                              hipStream_t stream) {
  (void)n_in; (void)out_size; (void)ws_size;

  const float* x   = (const float*)d_in[0];
  const int*   eil = (const int*)d_in[1];
  const int*   eig = (const int*)d_in[2];
  const float* P[24];
  for (int i = 0; i < 24; ++i) P[i] = (const float*)d_in[3 + i];
  const float* mW1 = P[16]; const float* mb1 = P[17];
  const float* mW2 = P[18]; const float* mb2 = P[19];
  const float* cb1 = P[21]; const float* cb2 = P[23];

  const int       N = in_sizes[0] / 256;     // 100000
  const long long E = in_sizes[1] / 2;       // 1000000
  const int INF = 256, H = 128, OUTF = 128, NCLS = 40, CH = 64;

  // ---- workspace layout: fp32 region, then f16 region ----
  float* ws    = (float*)d_ws;
  float* bufA  = ws;                          // N*128 fp32 (GEMM out -> propagate in)
  float* bufB  = bufA + (size_t)N * H;        // N*128 fp32 (aggregated features)
  float* dinvL = bufB + (size_t)N * H;        // N
  float* dinvG = dinvL + N;                   // N (contiguous with dinvL)
  float* stats = dinvG + N;                   // 256 (sum | sumsq)
  float* csum  = stats + 256;                 // 128 (+ pad to keep f16 aligned)

  _Float16* f16base = (_Float16*)(csum + 128 + 32);   // 32B-aligned f16 region
  _Float16* wtW1[2], *wtW2[2];
  wtW1[0] = f16base;                // 256*128
  wtW2[0] = wtW1[0] + 256 * 128;    // 128*128
  wtW1[1] = wtW2[0] + 128 * 128;
  wtW2[1] = wtW1[1] + 256 * 128;
  _Float16* wtM1 = wtW2[1] + 128 * 128;   // 128*128
  _Float16* wtM2 = wtM1 + 128 * 128;      // 128*128
  _Float16* wtC1 = wtM2 + 128 * 128;      // 128*64
  _Float16* wtC2 = wtC1 + 128 * 64;       // 64*40
  _Float16* xh16  = wtC2 + 64 * 40;             // N*256 (x in f16, both encoders)
  _Float16* bufB16 = xh16 + (size_t)N * INF;    // N*128 (BN out / classifier hidden)
  _Float16* bufC16 = bufB16 + (size_t)N * H;    // N*128 (MLP hidden / z_l+z_g)

  float* z_l  = (float*)d_out;
  float* z_g  = z_l + (size_t)N * OUTF;
  float* g_l  = z_g + (size_t)N * OUTF;
  float* g_g  = g_l + OUTF;
  float* pred = g_g + OUTF;

  // ---- pre-transpose all weights to f16 column-major; convert x to f16 ----
  transpose_f16_kernel<<<blk(INF * H, 256), 256, 0, stream>>>(P[0],  wtW1[0], INF, H);
  transpose_f16_kernel<<<blk(H * OUTF, 256), 256, 0, stream>>>(P[2],  wtW2[0], H, OUTF);
  transpose_f16_kernel<<<blk(INF * H, 256), 256, 0, stream>>>(P[8],  wtW1[1], INF, H);
  transpose_f16_kernel<<<blk(H * OUTF, 256), 256, 0, stream>>>(P[10], wtW2[1], H, OUTF);
  transpose_f16_kernel<<<blk(OUTF * OUTF, 256), 256, 0, stream>>>(mW1,   wtM1, OUTF, OUTF);
  transpose_f16_kernel<<<blk(OUTF * OUTF, 256), 256, 0, stream>>>(mW2,   wtM2, OUTF, OUTF);
  transpose_f16_kernel<<<blk(OUTF * CH, 256), 256, 0, stream>>>(P[20],  wtC1, OUTF, CH);
  transpose_f16_kernel<<<blk(CH * NCLS, 256), 256, 0, stream>>>(P[22],  wtC2, CH, NCLS);
  cvt16_kernel<<<blk((long long)N * INF, 256), 256, 0, stream>>>(x, xh16, (long long)N * INF);

  // ---- degree -> d^{-1/2} for both edge lists ----
  zero_kernel<<<blk(2LL * N, 256), 256, 0, stream>>>(dinvL, 2LL * N);
  deg_kernel<<<blk(E, 256), 256, 0, stream>>>(eil + E, dinvL, E);
  deg_kernel<<<blk(E, 256), 256, 0, stream>>>(eig + E, dinvG, E);
  rsqrt_kernel<<<blk(2LL * N, 256), 256, 0, stream>>>(dinvL, 2LL * N);

  const long long NH = (long long)N * H;
  const float invN = 1.0f / (float)N;

  for (int enc = 0; enc < 2; ++enc) {
    const float* const* p   = &P[enc * 8];     // W1 b1 W2 b2 g1 bt1 g2 bt2
    const int*   src  = (enc == 0) ? eil : eig;
    const int*   dst  = src + E;
    const float* dinv = (enc == 0) ? dinvL : dinvG;
    float* zout = (enc == 0) ? z_l : z_g;
    float* gout = (enc == 0) ? g_l : g_g;

    // GCN layer 1: t = x@W1 ; h = scatter(norm * t[src]) + b1 ; relu ; BN
    launch_gemm(xh16, wtW1[enc], nullptr, bufA, N, INF, H, 0, 0, stream);
    fill_bias_kernel<<<blk(NH, 256), 256, 0, stream>>>(bufB, p[1], NH, H - 1);
    propagate_kernel<<<blk(E * 32, 256), 256, 0, stream>>>(bufA, src, dst, dinv, bufB, E);
    relu_kernel<<<blk(NH, 256), 256, 0, stream>>>(bufB, NH);
    zero_kernel<<<1, 256, 0, stream>>>(stats, 256);
    col_stats_kernel<<<blk(N, 256), 128, 0, stream>>>(bufB, stats, N);
    bn16_kernel<<<blk(NH, 256), 256, 0, stream>>>(bufB, stats, p[4], p[5], bufB16, NH, invN);

    // GCN layer 2: t = h@W2 ; h2 = scatter + b2 ; BN
    launch_gemm(bufB16, wtW2[enc], nullptr, bufA, N, H, OUTF, 0, 0, stream);
    fill_bias_kernel<<<blk(NH, 256), 256, 0, stream>>>(bufB, p[3], NH, OUTF - 1);
    propagate_kernel<<<blk(E * 32, 256), 256, 0, stream>>>(bufA, src, dst, dinv, bufB, E);
    zero_kernel<<<1, 256, 0, stream>>>(stats, 256);
    col_stats_kernel<<<blk(N, 256), 128, 0, stream>>>(bufB, stats, N);
    bn16_kernel<<<blk(NH, 256), 256, 0, stream>>>(bufB, stats, p[6], p[7], bufB16, NH, invN);

    // projection MLP: z = relu(h@mW1 + mb1)@mW2 + mb2
    launch_gemm(bufB16, wtM1, mb1, bufC16, N, OUTF, OUTF, 1, 1, stream);
    launch_gemm(bufC16, wtM2, mb2, zout, N, OUTF, OUTF, 0, 0, stream);

    // graph summary: g = MLP(sum_rows(z))
    zero_kernel<<<1, 128, 0, stream>>>(csum, 128);
    colsum_kernel<<<blk(N, 256), 128, 0, stream>>>(zout, csum, N);
    tiny_mlp_kernel<<<1, 128, 0, stream>>>(csum, mW1, mb1, mW2, mb2, gout);
  }

  // node classifier: pred = relu((z_l + z_g)@cW1 + cb1)@cW2 + cb2
  add16_kernel<<<blk(NH, 256), 256, 0, stream>>>(z_l, z_g, bufC16, NH);
  launch_gemm(bufC16, wtC1, cb1, bufB16, N, OUTF, CH, 1, 1, stream);  // N*64 f16
  launch_gemm(bufB16, wtC2, cb2, pred, N, CH, NCLS, 0, 0, stream);
}
